// Conv2d_24713241822025
// MI455X (gfx1250) — matrix operand order repaired and checked by reading, not compiled, not run on hardware
//
#include <hip/hip_runtime.h>

typedef __attribute__((ext_vector_type(16))) __bf16 v16bf;
typedef __attribute__((ext_vector_type(8)))  float  v8f;
typedef unsigned short u16;

#define CIN    128
#define COUT   128
#define HDIM   128
#define WDIM   128
#define KHH    31
#define KWW    31
#define OHH    98
#define OWW    98
#define BATCH  16
#define NTOT   (BATCH*OHH*OWW)      /* 153664 */
#define KTOT   (CIN*KHH*KWW)        /* 123008 */
#define KBLOCKS (KTOT/32)           /* 3844 packed 32-wide blocks */
#define NSTEPS  (KTOT/64)           /* 1922 staged 64-wide steps  */
#define APACK_ELEMS ((size_t)KBLOCKS*4096)
#define XN_OFF_BYTES ((size_t)32<<20)
#define ROWU   72                   /* ushorts per LDS row = 144 B (128 B data + 16 B pad) */

__device__ __forceinline__ u16 f2bf(float f){
  unsigned u = __float_as_uint(f);
  u += 0x7fffu + ((u >> 16) & 1u);          // round-to-nearest-even
  return (u16)(u >> 16);
}

// ---- prep 1: filter fp32 -> bf16, packed per 32-wide K block: Apack[kblock][co][klocal]
__global__ __launch_bounds__(256) void pack_filter_k(const float* __restrict__ filt,
                                                     u16* __restrict__ Apack){
  size_t t = (size_t)blockIdx.x*256 + threadIdx.x;
  if (t >= APACK_ELEMS) return;
  int kblock = (int)(t >> 12);
  int r      = (int)(t & 4095);
  int co     = r >> 5;
  int klocal = r & 31;
  int k  = (kblock << 5) + klocal;          // k = (kh*KW + kw)*CIN + ci
  int ci = k & (CIN-1);
  int p  = k >> 7;
  int kh = p / KWW;
  int kw = p - kh*KWW;
  float v = filt[(((size_t)co*CIN + ci)*KHH + kh)*KWW + kw];
  Apack[t] = f2bf(v);
}

// ---- prep 2: x NCHW fp32 -> NHWC bf16
__global__ __launch_bounds__(256) void convert_x_k(const float* __restrict__ x,
                                                   u16* __restrict__ Xn){
  size_t t = (size_t)blockIdx.x*256 + threadIdx.x;
  if (t >= (size_t)BATCH*CIN*HDIM*WDIM) return;
  int w  = (int)(t & 127);
  int h  = (int)((t >> 7)  & 127);
  int ci = (int)((t >> 14) & 127);
  int b  = (int)(t >> 21);
  Xn[(((size_t)b*HDIM + h)*WDIM + w)*CIN + ci] = f2bf(x[t]);
}

// ---- main: implicit-GEMM conv, 128(COUT) x 128(pixels) per WG, bf16 WMMA, K staged 64/step
__global__ __launch_bounds__(256) void conv_wmma_k(const u16* __restrict__ Apack,
                                                   const u16* __restrict__ Xn,
                                                   const float* __restrict__ bias,
                                                   float* __restrict__ out){
  __shared__ u16 ldsA[2][128*ROWU];
  __shared__ u16 ldsB[2][128*ROWU];

  const int t    = threadIdx.x;
  const int lane = t & 31;
  const int wave = t >> 5;
  const int n0   = blockIdx.x * 128;

  // staging mapping (shared by A and B): 2 threads per row/column, 32 bf16 (64 B) each
  const int row  = t >> 1;
  const int half = t & 1;

  int nb = n0 + row; if (nb >= NTOT) nb = NTOT - 1;
  const int bb   = nb / (OHH*OWW);
  const int rem  = nb - bb*(OHH*OWW);
  const int ohp  = rem / OWW;
  const int owp  = rem - ohp*OWW;
  const u16* xcol = Xn + (((size_t)bb*HDIM + ohp)*WDIM + owp)*CIN;

  union FragAB { int4 i2[2]; v16bf v; };
  union FragC  { v8f  v;     float f[8]; };

  FragC acc[8];
  v8f zero = {0.f,0.f,0.f,0.f,0.f,0.f,0.f,0.f};
#pragma unroll
  for (int m = 0; m < 8; ++m) acc[m].v = zero;

  // prologue global loads for step 0 (kh=kw=0)
  int kh = 0, kw = 0;
  int4 aR[4], bR[4];
  {
    const int4* gA = (const int4*)(Apack + (size_t)half*4096 + (size_t)row*32);
    const int4* gB = (const int4*)(xcol + half*32);
#pragma unroll
    for (int i = 0; i < 4; ++i){ aR[i] = gA[i]; bR[i] = gB[i]; }
  }

  const int l15 = lane & 15, lh = lane >> 4;

  for (int s = 0; s < NSTEPS; ++s){
    const int buf = s & 1;

    // stage current 128x64 tiles into LDS
    {
      int4* sA = (int4*)(&ldsA[buf][row*ROWU + half*32]);
      int4* sB = (int4*)(&ldsB[buf][row*ROWU + half*32]);
#pragma unroll
      for (int i = 0; i < 4; ++i){ sA[i] = aR[i]; sB[i] = bR[i]; }
    }
    __syncthreads();

    // register-prefetch next step's tiles (overlaps WMMA below)
    if (s + 1 < NSTEPS){
      const int sn = s + 1;
      if ((sn & 1) == 0){ kw++; if (kw == KWW){ kw = 0; kh++; } }
      const int4* gA2 = (const int4*)(Apack + (size_t)(2*sn + half)*4096 + (size_t)row*32);
      const u16* px   = xcol + kh*(WDIM*CIN) + kw*CIN + (sn & 1)*64 + half*32;
      const int4* gB2 = (const int4*)px;
#pragma unroll
      for (int i = 0; i < 4; ++i){ aR[i] = gA2[i]; bR[i] = gB2[i]; }
    }

    // compute: two 32-wide K chunks
#pragma unroll
    for (int kc = 0; kc < 2; ++kc){
      // B fragment: column = wave*16 + (lane%16); 16 consecutive K per lane half
      FragAB bfr;
      const u16* pB = &ldsB[buf][(wave*16 + l15)*ROWU + kc*32 + lh*16];
      bfr.i2[0] = *(const int4*)(pB);
      bfr.i2[1] = *(const int4*)(pB + 8);
#pragma unroll
      for (int m = 0; m < 8; ++m){
        // A fragment: row = m*16 + (lane%16); 16 consecutive K per lane half
        FragAB afr;
        const u16* pA = &ldsA[buf][(m*16 + l15)*ROWU + kc*32 + lh*16];
        afr.i2[0] = *(const int4*)(pA);
        afr.i2[1] = *(const int4*)(pA + 8);
        acc[m].v = __builtin_amdgcn_wmma_f32_16x16x32_bf16(false, afr.v, false, bfr.v,
                                                           (short)0, acc[m].v, false, false);
      }
    }
    // double-buffered: single barrier per step is sufficient
  }

  // epilogue: C layout — VGPR r: lanes0-15 M=r, lanes16-31 M=8+r; N = lane%16
  const int nc = n0 + wave*16 + l15;
  if (nc < NTOT){
    const int b2  = nc / (OHH*OWW);
    const int r2  = nc - b2*(OHH*OWW);
    const int oh2 = r2 / OWW;
    const int ow2 = r2 - oh2*OWW;
    const int rowSel = lh * 8;
#pragma unroll
    for (int m = 0; m < 8; ++m){
#pragma unroll
      for (int r = 0; r < 8; ++r){
        const int co = m*16 + rowSel + r;
        out[(((size_t)b2*COUT + co)*OHH + oh2)*OWW + ow2] = acc[m].f[r] + bias[co];
      }
    }
  }
}

extern "C" void kernel_launch(void* const* d_in, const int* in_sizes, int n_in,
                              void* d_out, int out_size, void* d_ws, size_t ws_size,
                              hipStream_t stream){
  const float* x    = (const float*)d_in[0];
  const float* filt = (const float*)d_in[1];
  const float* bias = (const float*)d_in[2];
  float* out = (float*)d_out;

  u16* Apack = (u16*)d_ws;                                  // 31.5 MB
  u16* Xn    = (u16*)((char*)d_ws + XN_OFF_BYTES);          // 64 MB @ +32MB

  {
    size_t total = APACK_ELEMS;
    int blocks = (int)((total + 255)/256);
    pack_filter_k<<<blocks, 256, 0, stream>>>(filt, Apack);
  }
  {
    size_t total = (size_t)BATCH*CIN*HDIM*WDIM;
    int blocks = (int)((total + 255)/256);
    convert_x_k<<<blocks, 256, 0, stream>>>(x, Xn);
  }
  {
    int blocks = (NTOT + 127)/128;  // 1201
    conv_wmma_k<<<blocks, 256, 0, stream>>>(Apack, Xn, bias, out);
  }
}
